// GraphSageNet2_83073257439660
// MI455X (gfx1250) — matrix operand order repaired and checked by reading, not compiled, not run on hardware
//
#include <hip/hip_runtime.h>
#include <hip/hip_bf16.h>
#include <cstdint>
#include <cstddef>

// ---------------- problem constants (fixed by the reference) ----------------
#define N_NODES  100000
#define N_EDGES  1600000
#define N_GRAPHS 128
#define DH       108       // D
#define IN_DIM   146
#define N_LAYERS 4

typedef _Float16 v8h  __attribute__((ext_vector_type(8)));
typedef _Float16 v16h __attribute__((ext_vector_type(16)));
typedef float    v8f  __attribute__((ext_vector_type(8)));

// ---- WMMA fragment loaders (layouts per CDNA5 ISA 7.12.2, wave32) ----------
// A (16x32 f16): lanes 0-15 row=lane hold K {0..7,16..23}; lanes 16-31 row=lane-16
// hold K {8..15,24..31}.  Two 16B LDS loads per lane.
__device__ __forceinline__ v16h frag_A(const _Float16* rowp, int kbase, int hi16) {
  const _Float16* p = rowp + kbase + (hi16 << 3);
  v8h lo = *(const v8h*)(p);
  v8h hi = *(const v8h*)(p + 16);
  return __builtin_shufflevector(lo, hi, 0,1,2,3,4,5,6,7,8,9,10,11,12,13,14,15);
}
// B (32x16 f16): col = lane&15; lanes 0-15 hold K {0..7,16..23}; lanes 16-31 hold
// K {8..15,24..31} (same order as A). Two 16B LDS loads per lane.
__device__ __forceinline__ v16h frag_B(const _Float16* colp, int kbase, int hi16) {
  const _Float16* p = colp + kbase + (hi16 << 3);
  v8h lo = *(const v8h*)(p);
  v8h hi = *(const v8h*)(p + 16);
  return __builtin_shufflevector(lo, hi, 0,1,2,3,4,5,6,7,8,9,10,11,12,13,14,15);
}

// ---------------------------------------------------------------------------
// Fused GEMM: Y[r][n] = act( sum_k A(r,k) * W[n][k] + b[n] )
// A(r,k) = X0[r][k] for k<K0, else X1[r][k-K0] * (1/max(deg[r],1)) (node layer
// concat with mean-agg scaling folded in).  f16 staging in LDS (8-wide packed
// ds_store_b128), f32 accumulate via v_wmma_f32_16x16x32_f16.
// 256 thr = 8 waves; 128 rows/block; NoP = 112 fixed; KP compile-time mult of 32.
// A fragments preloaded to registers; B fragments double-buffered one tile
// ahead so ds_load latency overlaps the XDL WMMA pipeline.
// ---------------------------------------------------------------------------
template <int KP>
__global__ void k_gemm_wmma(const float* __restrict__ X0, int K0,
                            const float* __restrict__ X1, int K1,
                            const float* __restrict__ deg,
                            const float* __restrict__ W,
                            const float* __restrict__ b,
                            float* __restrict__ Y,
                            int Nrows, int No, int relu) {
  extern __shared__ char smem[];
  _Float16* Ah = (_Float16*)smem;            // [128][KP]
  _Float16* Wh = Ah + 128 * KP;              // [112][KP]
  float*    Bb = (float*)(Wh + 112 * KP);    // [112] zero-padded bias
  const int tid = threadIdx.x;
  const int rowBase = blockIdx.x * 128;
  const int K = K0 + K1;
  constexpr int CPR = KP / 8;                // 8-half chunks per row

  // ---- stage A tile: f32 -> f16, zero padded, one ds_store_b128 per chunk
  for (int idx = tid; idx < 128 * CPR; idx += 256) {
    const int row = idx / CPR;               // constant divisor -> cheap
    const int kb  = (idx - row * CPR) * 8;
    const int gr  = rowBase + row;
    v8h pack = {};
    if (gr < Nrows) {
      const float s = deg ? (1.f / fmaxf(deg[gr], 1.f)) : 1.f;
      const float* r0 = X0 + (size_t)gr * K0;
      const float* r1 = X1 ? (X1 + (size_t)gr * K1) : nullptr;
#pragma unroll
      for (int i = 0; i < 8; ++i) {
        const int k = kb + i;
        float v = 0.f;
        if (k < K0)      v = r0[k];
        else if (k < K)  v = r1[k - K0] * s;
        pack[i] = (_Float16)v;
      }
    }
    *(v8h*)(Ah + row * KP + kb) = pack;
  }
  // ---- stage W (row n = output neuron n, stride K), zero padded
  for (int idx = tid; idx < 112 * CPR; idx += 256) {
    const int n  = idx / CPR;
    const int kb = (idx - n * CPR) * 8;
    v8h pack = {};
    if (n < No) {
      const float* wr = W + (size_t)n * K;
#pragma unroll
      for (int i = 0; i < 8; ++i) {
        const int k = kb + i;
        pack[i] = (_Float16)((k < K) ? wr[k] : 0.f);
      }
    }
    *(v8h*)(Wh + n * KP + kb) = pack;
  }
  // ---- stage zero-padded bias
  if (tid < 112) Bb[tid] = (tid < No) ? b[tid] : 0.f;
  __syncthreads();

  const int lane = tid & 31;
  const int wv   = tid >> 5;            // wave id: rows wv*16 .. wv*16+15
  const int l15  = lane & 15;
  const int hi16 = lane >> 4;
  const _Float16* Arow = Ah + (size_t)(wv * 16 + l15) * KP;
  constexpr int KT = KP / 32;
  const int NT = 112 >> 4;              // 7

  // A fragments are invariant across N-tiles: keep them in registers.
  v16h av[KT];
#pragma unroll
  for (int kt = 0; kt < KT; ++kt) av[kt] = frag_A(Arow, kt * 32, hi16);

  const bool fullTile = (rowBase + 128 <= Nrows);
  const int grBase = rowBase + wv * 16 + (hi16 << 3);

  for (int nt = 0; nt < NT; ++nt) {
    const int n = nt * 16 + l15;        // < 112 always
    const float bv = Bb[n];
    v8f acc = {bv, bv, bv, bv, bv, bv, bv, bv};
    const _Float16* Bcol = Wh + (size_t)n * KP;
    // one-tile-ahead B double buffer: loads for kt+1 in flight during wmma kt
    v16h bcur = frag_B(Bcol, 0, hi16);
#pragma unroll
    for (int kt = 0; kt < KT; ++kt) {
      v16h bnxt = (kt + 1 < KT) ? frag_B(Bcol, (kt + 1) * 32, hi16) : bcur;
      acc = __builtin_amdgcn_wmma_f32_16x16x32_f16(
          false, av[kt], false, bcur, (short)0, acc, false, false);
      bcur = bnxt;
    }
    if (n < No) {
      // C/D layout: VGPR j -> M = j + 8*hi16, N = lane&15
      float* yp = Y + (size_t)grBase * No + n;
      if (fullTile) {
#pragma unroll
        for (int j = 0; j < 8; ++j) {
          float o = acc[j];
          if (relu) o = fmaxf(o, 0.f);
          yp[(size_t)j * No] = o;
        }
      } else {
#pragma unroll
        for (int j = 0; j < 8; ++j) {
          if (grBase + j < Nrows) {
            float o = acc[j];
            if (relu) o = fmaxf(o, 0.f);
            yp[(size_t)j * No] = o;
          }
        }
      }
    }
  }
}

// -------- in-degree: deg[d] += 1 per edge (hardware f32 atomic, L2) ---------
__global__ void k_deg(const int* __restrict__ dst, float* __restrict__ deg) {
  int e = blockIdx.x * blockDim.x + threadIdx.x;
  if (e >= N_EDGES) return;
  unsafeAtomicAdd(&deg[dst[e]], 1.0f);
}

// -------- agg[dst] += pooled[src] : thread per (edge, 4-float chunk) --------
__global__ void k_edge_agg(const int* __restrict__ src, const int* __restrict__ dst,
                           const float* __restrict__ pooled, float* __restrict__ agg) {
  const int CH = DH / 4;  // 27 float4 chunks per row (108*4B row is 16B aligned)
  long long idx = (long long)blockIdx.x * blockDim.x + threadIdx.x;
  if (idx >= (long long)N_EDGES * CH) return;
  int e = (int)(idx / CH), c = (int)(idx - (long long)e * CH);
  int s = src[e], d = dst[e];
  const float4 v = *(const float4*)(pooled + (size_t)s * DH + c * 4);
  float* p = agg + (size_t)d * DH + c * 4;
  unsafeAtomicAdd(p + 0, v.x);
  unsafeAtomicAdd(p + 1, v.y);
  unsafeAtomicAdd(p + 2, v.z);
  unsafeAtomicAdd(p + 3, v.w);
}

// ---- NodeApply tail: h = h + relu(tmp / max(||tmp||,1e-12)) * snorm --------
// one wave32 per row; cross-lane sum via __shfl_xor
__global__ void k_node_post(float* __restrict__ h, const float* __restrict__ tmp,
                            const float* __restrict__ snorm) {
  int row = blockIdx.x * (blockDim.x >> 5) + (threadIdx.x >> 5);
  if (row >= N_NODES) return;
  int lane = threadIdx.x & 31;
  float vals[4];
  float ss = 0.f;
#pragma unroll
  for (int i = 0; i < 4; ++i) {
    int j = lane + i * 32;
    float v = (j < DH) ? tmp[(size_t)row * DH + j] : 0.f;
    vals[i] = v;
    ss += v * v;
  }
#pragma unroll
  for (int off = 16; off > 0; off >>= 1) ss += __shfl_xor(ss, off, 32);
  float r = fmaxf(sqrtf(ss), 1e-12f);
  float scale = snorm[row] / r;
#pragma unroll
  for (int i = 0; i < 4; ++i) {
    int j = lane + i * 32;
    if (j < DH) {
      size_t ix = (size_t)row * DH + j;
      h[ix] += fmaxf(vals[i], 0.f) * scale;
    }
  }
}

// -------- per-graph mean readout: atomic sums + counts ----------------------
__global__ void k_readout(const float* __restrict__ h, const int* __restrict__ gid,
                          float* __restrict__ hg, float* __restrict__ cnt) {
  long long idx = (long long)blockIdx.x * blockDim.x + threadIdx.x;
  if (idx >= (long long)N_NODES * DH) return;
  int nrow = (int)(idx / DH), j = (int)(idx - (long long)nrow * DH);
  int g = gid[nrow];
  unsafeAtomicAdd(&hg[(size_t)g * DH + j], h[idx]);
  if (j == 0) unsafeAtomicAdd(&cnt[g], 1.0f);
}

// -------- tiny MLP readout: 108 -> 54 -> 27 -> 10, one thread per graph -----
__global__ void k_mlp(const float* __restrict__ hg, const float* __restrict__ cnt,
                      const float* __restrict__ W0, const float* __restrict__ b0,
                      const float* __restrict__ W1, const float* __restrict__ b1,
                      const float* __restrict__ W2, const float* __restrict__ b2,
                      float* __restrict__ out) {
  int g = blockIdx.x * blockDim.x + threadIdx.x;
  if (g >= N_GRAPHS) return;
  float x[DH];
  float c = fmaxf(cnt[g], 1.f);
  for (int j = 0; j < DH; ++j) x[j] = hg[(size_t)g * DH + j] / c;
  float y[DH / 2];
  for (int o = 0; o < DH / 2; ++o) {
    float s = b0[o];
    for (int j = 0; j < DH; ++j) s += W0[(size_t)o * DH + j] * x[j];
    y[o] = fmaxf(s, 0.f);
  }
  float z[DH / 4];
  for (int o = 0; o < DH / 4; ++o) {
    float s = b1[o];
    for (int j = 0; j < DH / 2; ++j) s += W1[(size_t)o * (DH / 2) + j] * y[j];
    z[o] = fmaxf(s, 0.f);
  }
  for (int o = 0; o < 10; ++o) {
    float s = b2[o];
    for (int j = 0; j < DH / 4; ++j) s += W2[(size_t)o * (DH / 4) + j] * z[j];
    out[(size_t)g * 10 + o] = s;
  }
}

// ---------------------------------------------------------------------------
extern "C" void kernel_launch(void* const* d_in, const int* in_sizes, int n_in,
                              void* d_out, int out_size, void* d_ws, size_t ws_size,
                              hipStream_t stream) {
  const float* nodes_feat = (const float*)d_in[0];
  const float* snorm      = (const float*)d_in[1];
  const int*   src        = (const int*)d_in[2];
  const int*   dst        = (const int*)d_in[3];
  const int*   gid        = (const int*)d_in[4];
  // d_in[5] = num_graphs (128, compile-time constant here)
  const float* W_emb  = (const float*)d_in[6];
  const float* b_emb  = (const float*)d_in[7];
  const float* W_pool = (const float*)d_in[8];
  const float* b_pool = (const float*)d_in[9];
  const float* W_node = (const float*)d_in[10];
  const float* b_node = (const float*)d_in[11];
  const float* W_r0 = (const float*)d_in[12];
  const float* b_r0 = (const float*)d_in[13];
  const float* W_r1 = (const float*)d_in[14];
  const float* b_r1 = (const float*)d_in[15];
  const float* W_r2 = (const float*)d_in[16];
  const float* b_r2 = (const float*)d_in[17];
  float* out = (float*)d_out;

  // workspace carve-out (~130 MB); pooled doubles as the node-GEMM temp
  char* ws = (char*)d_ws;
  auto carve = [&](size_t bytes) -> char* {
    char* p = ws;
    ws += (bytes + 255) & ~(size_t)255;
    return p;
  };
  float* h      = (float*)carve((size_t)N_NODES * DH * 4);
  float* pooled = (float*)carve((size_t)N_NODES * DH * 4);  // also "tmp"
  float* agg    = (float*)carve((size_t)N_NODES * DH * 4);
  float* deg    = (float*)carve((size_t)N_NODES * 4);
  float* hg     = (float*)carve((size_t)N_GRAPHS * DH * 4);
  float* cnt    = (float*)carve((size_t)N_GRAPHS * 4);

  const int gemmGrid = (N_NODES + 127) / 128;
  auto shmem = [](int KP) { return (size_t)(128 + 112) * KP * sizeof(_Float16) + 112 * sizeof(float); };

  // in-degree (fixed across layers)
  hipMemsetAsync(deg, 0, (size_t)N_NODES * 4, stream);
  k_deg<<<(N_EDGES + 255) / 256, 256, 0, stream>>>(dst, deg);

  // embedding: h = nodes_feat @ W_emb^T + b_emb   (K=146 -> KP=160)
  k_gemm_wmma<160><<<gemmGrid, 256, shmem(160), stream>>>(
      nodes_feat, IN_DIM, nullptr, 0, nullptr, W_emb, b_emb, h, N_NODES, DH, 0);

  for (int l = 0; l < N_LAYERS; ++l) {
    hipMemsetAsync(agg, 0, (size_t)N_NODES * DH * 4, stream);
    // pooled = relu(h @ W_pool[l]^T + b)   (K=108 -> KP=128)
    k_gemm_wmma<128><<<gemmGrid, 256, shmem(128), stream>>>(
        h, DH, nullptr, 0, nullptr, W_pool + (size_t)l * DH * DH,
        b_pool + (size_t)l * DH, pooled, N_NODES, DH, 1);
    // agg[dst] += pooled[src]
    {
      long long tot = (long long)N_EDGES * (DH / 4);
      k_edge_agg<<<(unsigned)((tot + 255) / 256), 256, 0, stream>>>(src, dst,
                                                                    pooled, agg);
    }
    // tmp = concat(h, agg/deg) @ W_node[l]^T + b   (K=216 -> KP=224)
    k_gemm_wmma<224><<<gemmGrid, 256, shmem(224), stream>>>(
        h, DH, agg, DH, deg, W_node + (size_t)l * DH * 2 * DH,
        b_node + (size_t)l * DH, pooled /*tmp*/, N_NODES, DH, 0);
    // h = h + relu(normalize(tmp)) * snorm
    k_node_post<<<(N_NODES + 7) / 8, 256, 0, stream>>>(h, pooled, snorm);
  }

  // per-graph mean + MLP readout
  hipMemsetAsync(hg, 0, (size_t)N_GRAPHS * DH * 4, stream);
  hipMemsetAsync(cnt, 0, (size_t)N_GRAPHS * 4, stream);
  {
    long long tot = (long long)N_NODES * DH;
    k_readout<<<(unsigned)((tot + 255) / 256), 256, 0, stream>>>(h, gid, hg, cnt);
  }
  k_mlp<<<1, 128, 0, stream>>>(hg, cnt, W_r0, b_r0, W_r1, b_r1, W_r2, b_r2, out);
}
